// conv1d_17471926960393
// MI455X (gfx1250) — compile-verified, matrix operand order checked by reading, not run on hardware
//
#include <hip/hip_runtime.h>
#include <math.h>

typedef __attribute__((ext_vector_type(16))) _Float16 v16h;
typedef __attribute__((ext_vector_type(8)))  float    v8f;

#define NCLS 21
#define KW 16
#define KPAD 22                 // classes padded 21 -> 22 so K is a multiple of 32
#define KTOT (KW * KPAD)        // 352
#define KSTEPS (KTOT / 32)      // 11
#define LPOS 985
#define LIN 1000
#define BATCH 4096
#define ROWS_PER_WG 128
#define WAVES_PER_WG 8
#define NUM_TILES ((BATCH * LPOS) / ROWS_PER_WG)   // 31,520
#define GRID_BLOCKS 3940                           // 8 tiles per persistent block

// ISA 7.12.2: 16-bit A/B matrix K index for fragment element e (= 2*v + h)
// lanes 0-15 vs 16-31 split K ranges; VGPRs 0-3 cover K 0..15, 4-7 cover 16..31.
__device__ __forceinline__ int klocal(int e, int laneHi) {
    int v = e >> 1, h = e & 1;
    return 2 * v + (v >= 4 ? 8 : 0) + (laneHi ? 8 : 0) + h;
}

// gfx1250 async global->LDS copy, GVS (saddr) form, tracked by ASYNCcnt.
// LDS destination address = low 32 bits of the generic pointer to __shared__
// (flat->LDS truncation, ISA 10.2).
__device__ __forceinline__ void async_g2l_b32_s(void* lds, const void* base,
                                                unsigned byteoff) {
    unsigned ldsoff = (unsigned)(unsigned long long)lds;
    asm volatile("global_load_async_to_lds_b32 %0, %1, %2"
                 :: "v"(ldsoff), "v"(byteoff), "s"(base) : "memory");
}
__device__ __forceinline__ void wait_asynccnt0() {
    asm volatile("s_wait_asynccnt 0x0" ::: "memory");
}

// Issue async loads of one tile's per-row x windows (128 rows x 16 taps) into LDS.
__device__ __forceinline__ void stage_x_async(const int* __restrict__ x,
                                              unsigned R0, int tid, int* dst) {
    for (int i = tid; i < ROWS_PER_WG * 16; i += 256) {
        int rl = i >> 4, t = i & 15;
        unsigned R = R0 + (unsigned)rl;
        int b = (int)(R / LPOS);
        int p = (int)(R % LPOS);
        async_g2l_b32_s(&dst[i], x, (unsigned)((b * LIN + p + t) * 4));
    }
}

__global__ __launch_bounds__(256) void conv_onehot_wmma_kernel(
    const int* __restrict__ x, const float* __restrict__ w,
    const float* __restrict__ bias, const float* __restrict__ fcw,
    float* __restrict__ wsacc)
{
    __shared__ _Float16 sB[KSTEPS * 4 * 32 * 16];    // 45056 B, pre-swizzled B fragments
    __shared__ int      sX[2][ROWS_PER_WG * 16];     // 2 x 8192 B, double-buffered x windows
    __shared__ float    sY[WAVES_PER_WG * 16 * 64];  // 32768 B, per-wave conv tiles
    __shared__ float    sAcc[2];                     // per-WG batch accumulators

    const int tid    = threadIdx.x;
    const int lane   = tid & 31;
    const int wave   = tid >> 5;
    const int laneHi = lane >= 16;

    // ---- Stage B fragments ONCE per persistent block ----
    for (int i = tid; i < KSTEPS * 4 * 32 * 16; i += 256) {
        int e  = i & 15;
        int ln = (i >> 4) & 31;
        int nt = (i >> 9) & 3;
        int ks = i >> 11;
        int Kg  = ks * 32 + klocal(e, ln >= 16);
        int tap = Kg / KPAD;
        int c   = Kg % KPAD;
        int o   = nt * 16 + (ln & 15);
        _Float16 val = (_Float16)0.f;
        if (c < NCLS) val = (_Float16)w[(o * NCLS + c) * KW + tap];
        sB[i] = val;
    }
    if (tid < 2) sAcc[tid] = 0.f;

    // ---- Prologue: stage first tile's x windows ----
    unsigned blk = blockIdx.x;
    if (blk < NUM_TILES) stage_x_async(x, blk * ROWS_PER_WG, tid, sX[0]);
    wait_asynccnt0();
    __syncthreads();   // sB + sX[0] + sAcc visible to all waves

    // ---- Persistent grid-stride loop over 128-row M tiles ----
    int pb = 0;
    for (; blk < NUM_TILES; blk += gridDim.x, pb ^= 1) {
        const unsigned R0 = blk * ROWS_PER_WG;
        const int* xw = sX[pb];

        // Kick off NEXT tile's async x staging; it runs under this tile's WMMAs.
        unsigned nblk = blk + gridDim.x;
        if (nblk < NUM_TILES) stage_x_async(x, nblk * ROWS_PER_WG, tid, sX[pb ^ 1]);

        // ---- Per-wave GEMM: 16 rows x (K=352) x 64 channels ----
        const int mrow    = lane & 15;
        const int rl_base = wave * 16;
        {   // prefetch the fc_w slice this wave consumes in the epilogue
            unsigned R = R0 + (unsigned)(rl_base + mrow);
            int p = (int)(R % LPOS);
            __builtin_prefetch(&fcw[p * 12], 0, 0);
        }

        v8f acc0 = {}, acc1 = {}, acc2 = {}, acc3 = {};
        for (int ks = 0; ks < KSTEPS; ++ks) {
            // Build the one-hot A fragment in registers (1 cndmask per element)
            v16h a;
            #pragma unroll
            for (int e = 0; e < 16; ++e) {
                int Kg  = ks * 32 + klocal(e, laneHi);
                int tap = Kg / KPAD;
                int c   = Kg % KPAD;
                int xv  = xw[(rl_base + mrow) * 16 + tap];
                a[e] = (c < NCLS && xv == c) ? (_Float16)1.f : (_Float16)0.f;
            }
            const _Float16* bb = &sB[(ks * 4) * 32 * 16 + lane * 16];
            v16h b0 = *(const v16h*)(bb + 0 * 32 * 16);
            v16h b1 = *(const v16h*)(bb + 1 * 32 * 16);
            v16h b2 = *(const v16h*)(bb + 2 * 32 * 16);
            v16h b3 = *(const v16h*)(bb + 3 * 32 * 16);
            acc0 = __builtin_amdgcn_wmma_f32_16x16x32_f16(false, a, false, b0, (short)0, acc0, false, false);
            acc1 = __builtin_amdgcn_wmma_f32_16x16x32_f16(false, a, false, b1, (short)0, acc1, false, false);
            acc2 = __builtin_amdgcn_wmma_f32_16x16x32_f16(false, a, false, b2, (short)0, acc2, false, false);
            acc3 = __builtin_amdgcn_wmma_f32_16x16x32_f16(false, a, false, b3, (short)0, acc3, false, false);
        }

        // ---- Epilogue: bias + ReLU into per-wave LDS tile (row-major 16x64) ----
        float* yw = &sY[wave * 16 * 64];
        #pragma unroll
        for (int nt = 0; nt < 4; ++nt) {
            v8f cc = (nt == 0) ? acc0 : (nt == 1) ? acc1 : (nt == 2) ? acc2 : acc3;
            int o = nt * 16 + (lane & 15);
            float bo = bias[o];
            #pragma unroll
            for (int j = 0; j < 8; ++j) {
                int row = j + (laneHi ? 8 : 0);
                float v = cc[j] + bo;
                yw[row * 64 + o] = v > 0.f ? v : 0.f;
            }
        }
        // Wave-local LDS ops are in order; this wave only reads its own y tile.

        // ---- Max-pool(5) over channels + dot with fc_w; per-row partials ----
        {
            int row = lane & 15;
            unsigned R = R0 + (unsigned)(rl_base + row);
            int b = (int)(R / LPOS);
            int p = (int)(R % LPOS);
            int g0 = laneHi ? 6 : 0;    // two lanes split the 12 groups per row
            float partial = 0.f;
            #pragma unroll
            for (int gg = 0; gg < 6; ++gg) {
                int g = g0 + gg;
                float m = yw[row * 64 + g * 5];
                #pragma unroll
                for (int t = 1; t < 5; ++t) m = fmaxf(m, yw[row * 64 + g * 5 + t]);
                partial = fmaf(m, fcw[p * 12 + g], partial);
            }
            int b0 = (int)(R0 / LPOS);  // WG tile spans at most 2 batch indices
            atomicAdd(&sAcc[b - b0], partial);     // ds_add_f32
        }
        __syncthreads();                // all pooling atomics done
        if (tid < 2) {
            int b0 = (int)(R0 / LPOS);
            if (b0 + tid < BATCH) atomicAdd(&wsacc[b0 + tid], sAcc[tid]);
            sAcc[tid] = 0.f;
        }
        wait_asynccnt0();               // own async stores for next tile complete
        __syncthreads();                // all waves' asyncs + sAcc reset visible
    }
}

__global__ void zero_acc_kernel(float* __restrict__ a, int n) {
    int i = blockIdx.x * 256 + threadIdx.x;
    if (i < n) a[i] = 0.f;
}

__global__ void sigmoid_kernel(const float* __restrict__ acc,
                               const float* __restrict__ fcb,
                               float* __restrict__ out, int n) {
    int i = blockIdx.x * 256 + threadIdx.x;
    if (i < n) {
        float z = acc[i] + fcb[0];
        out[i] = 1.f / (1.f + __expf(-z));
    }
}

extern "C" void kernel_launch(void* const* d_in, const int* in_sizes, int n_in,
                              void* d_out, int out_size, void* d_ws, size_t ws_size,
                              hipStream_t stream) {
    const int*   x    = (const int*)d_in[0];
    const float* w    = (const float*)d_in[1];
    const float* bias = (const float*)d_in[2];
    const float* fcw  = (const float*)d_in[3];
    const float* fcb  = (const float*)d_in[4];
    float* out   = (float*)d_out;
    float* wsacc = (float*)d_ws;

    zero_acc_kernel<<<(BATCH + 255) / 256, 256, 0, stream>>>(wsacc, BATCH);

    conv_onehot_wmma_kernel<<<GRID_BLOCKS, 256, 0, stream>>>(x, w, bias, fcw, wsacc);

    sigmoid_kernel<<<(BATCH + 255) / 256, 256, 0, stream>>>(wsacc, fcb, out, BATCH);
}
